// PointTransformerLast_4269197492668
// MI455X (gfx1250) — matrix operand order repaired and checked by reading, not compiled, not run on hardware
//
#include <hip/hip_runtime.h>
#include <hip/hip_bf16.h>
#include <stdint.h>

// Problem constants (from reference setup_inputs)
#define Bn 16
#define Cc 256
#define Nn 2048
#define Hh 64
#define NEGV (-9.0e15f)

static constexpr int BM_T = 64;   // GEMM M tile
static constexpr int BN_T = 128;  // GEMM N tile
static constexpr int BK_T = 32;   // GEMM K step (matches wmma bf16 K=32)

typedef __attribute__((ext_vector_type(16))) __bf16 v16bf;
typedef __attribute__((ext_vector_type(8)))  float  v8f;

union Frag { v16bf v; uint4 q[2]; };

__device__ __forceinline__ unsigned short f2bf(float f) {
  unsigned u = __builtin_bit_cast(unsigned, f);
  u += 0x7FFFu + ((u >> 16) & 1u);          // RNE
  return (unsigned short)(u >> 16);
}
__device__ __forceinline__ float bf2f(unsigned short h) {
  unsigned u = ((unsigned)h) << 16;
  return __builtin_bit_cast(float, u);
}
__device__ __forceinline__ v8f wmma_bf16(const Frag& a, const Frag& b, v8f c) {
  return __builtin_amdgcn_wmma_f32_16x16x32_bf16(
      false, a.v, false, b.v, (short)0, c, false, false);
}
// A fragment (16x32 bf16): per ISA 7.12.2 lanes 0-15 hold K {0..7,16..23},
// lanes 16-31 hold K {8..15,24..31} of rows M=lane&15.  Two b128 loads.
__device__ __forceinline__ void load_fragA(Frag& f, const unsigned short* row, int half) {
  f.q[0] = *(const uint4*)(row + 8 * half);
  f.q[1] = *(const uint4*)(row + 16 + 8 * half);
}
// B fragment (32x16 bf16): lanes 0-15 hold K {0..7,16..23}, lanes 16-31 K
// {8..15,24..31} of column N=lane&15 (same K order as A), B stored [col][k].
__device__ __forceinline__ void load_fragB(Frag& f, const unsigned short* colrow, int half) {
  f.q[0] = *(const uint4*)(colrow + 8 * half);
  f.q[1] = *(const uint4*)(colrow + 16 + 8 * half);
}

// CDNA5 async global->LDS copy (ASYNCcnt-tracked, bypasses VGPRs).
__device__ __forceinline__ void async_lds_b128(void* lds, const void* gsrc) {
  unsigned dst = (unsigned)(uintptr_t)lds;               // LDS byte offset (low 32)
  unsigned long long src = (unsigned long long)(uintptr_t)gsrc;
  asm volatile("global_load_async_to_lds_b128 %0, %1, off"
               :: "v"(dst), "v"(src) : "memory");
}
// ASYNC loads retire in order per wave: waiting "<= n" with n = copies issued
// for the NEXT tile guarantees the CURRENT tile's copies have landed.
__device__ __forceinline__ void async_wait0() {
  asm volatile("s_wait_asynccnt 0" ::: "memory");
}
__device__ __forceinline__ void async_wait1() {
  asm volatile("s_wait_asynccnt 1" ::: "memory");
}
__device__ __forceinline__ void async_wait3() {
  asm volatile("s_wait_asynccnt 3" ::: "memory");
}

// ---------------------------------------------------------------------------
// Generic GEMM: D[b][M][N] = A[M][K](bf16, shared over batch) * X[b][K][N](bf16)
// Double-buffered LDS; A-tile staged with pipelined async-to-LDS copies,
// B-tile software-pipelined through VGPRs (transpose scatter on the LDS side).
// EPI 0: store f32 row-major.  EPI 1: +bias, store bf16 row-major.
// EPI 2: store bf16 transposed ([b][n][M]) -- used to build qT for attention.
// ---------------------------------------------------------------------------
template <int EPI>
__global__ __launch_bounds__(256) void k_gemm(
    const unsigned short* __restrict__ A, const unsigned short* __restrict__ X,
    int M, int K, int Nt, const float* __restrict__ bias,
    float* __restrict__ outF, unsigned short* __restrict__ outB) {
  __shared__ __align__(16) unsigned short ldsA[2][BM_T * BK_T];   // [m][k]
  __shared__ __align__(16) unsigned short ldsBT[2][BN_T * BK_T];  // [n][k]
  const int b = blockIdx.z, n0 = blockIdx.x * BN_T, m0 = blockIdx.y * BM_T;
  const int tid = threadIdx.x, lane = tid & 31, w = tid >> 5;
  const int wm = w >> 1, wn = w & 1, half = lane >> 4, lj = lane & 15;
  const unsigned short* Xb = X + (size_t)b * K * Nt;

  // staging coordinates
  const int am = tid >> 2, akk = (tid & 3) * 8;    // A: 16B per thread
  const int bkk = tid >> 3, bcc = (tid & 7) * 16;  // B: 32B per thread
  const unsigned short* Arow = A + (size_t)(m0 + am) * K + akk;
  const unsigned short* Bsrc = Xb + (size_t)bkk * Nt + n0 + bcc;

  v8f acc[4];
  for (int t = 0; t < 4; ++t)
    for (int i = 0; i < 8; ++i) acc[t][i] = 0.0f;

  // prologue: tile 0 (A async, B into registers)
  async_lds_b128(ldsA[0] + am * BK_T + akk, Arow);
  uint4 breg0 = *(const uint4*)Bsrc;
  uint4 breg1 = *(const uint4*)(Bsrc + 8);

  for (int k0 = 0; k0 < K; k0 += BK_T) {
    const int buf = (k0 / BK_T) & 1;
    const bool more = (k0 + BK_T) < K;
    if (more) {
      async_lds_b128(ldsA[buf ^ 1] + am * BK_T + akk, Arow + k0 + BK_T);
      if (k0 + 2 * BK_T < K)  // global_prefetch_b8 two tiles ahead
        __builtin_prefetch(Bsrc + (size_t)(k0 + 2 * BK_T) * Nt, 0, 1);
    }
    {  // scatter current B registers into ldsBT[buf] (transposed [n][k])
      unsigned short tmp[16];
      *(uint4*)tmp = breg0; *(uint4*)(tmp + 8) = breg1;
      for (int i = 0; i < 16; ++i) ldsBT[buf][(bcc + i) * BK_T + bkk] = tmp[i];
    }
    if (more) {  // issue next B-tile global loads before the WMMAs
      const unsigned short* s = Bsrc + (size_t)(k0 + BK_T) * Nt;
      breg0 = *(const uint4*)s; breg1 = *(const uint4*)(s + 8);
    }
    if (more) async_wait1(); else async_wait0();
    __syncthreads();
    Frag fa; load_fragA(fa, ldsA[buf] + (wm * 16 + lj) * BK_T, half);
    for (int t = 0; t < 4; ++t) {
      Frag fb; load_fragB(fb, ldsBT[buf] + (wn * 64 + t * 16 + lj) * BK_T, half);
      acc[t] = wmma_bf16(fa, fb, acc[t]);
    }
    __syncthreads();
  }

  for (int t = 0; t < 4; ++t) {
    int col = n0 + wn * 64 + t * 16 + lj;
    int rbase = m0 + wm * 16 + 8 * half;  // D: VGPR r holds row r + 8*half
    if (EPI == 0) {
      for (int r = 0; r < 8; ++r)
        outF[((size_t)b * M + rbase + r) * Nt + col] = acc[t][r];
    } else if (EPI == 1) {
      for (int r = 0; r < 8; ++r)
        outB[((size_t)b * M + rbase + r) * Nt + col] = f2bf(acc[t][r] + bias[rbase + r]);
    } else {  // EPI==2: qT[b][n][h], 8 contiguous h per lane -> b128 store
      uint4 pk; unsigned short* tp = (unsigned short*)&pk;
      for (int r = 0; r < 8; ++r) tp[r] = f2bf(acc[t][r]);
      *(uint4*)(outB + ((size_t)b * Nt + col) * M + rbase) = pk;
    }
  }
}

// ---------------------------------------------------------------------------
// Attention energy + masked softmax (flash two-sweep), emits attnT bf16.
// S[n,m] = sum_h qT[n,h]*qT[m,h]; attnT[b][m][n] = softmax_row_n(S)[n,m].
// Each wave owns 16 rows n.  Sweep 1 keeps PER-LANE online (max,sum) pairs
// (each lane only ever sees columns m == lane (mod 16)); the 16-lane
// log-sum-exp merge happens once at the end.
// ---------------------------------------------------------------------------
__global__ __launch_bounds__(256) void k_energy(
    const unsigned short* __restrict__ qT,  // [B][N][H]
    const int* __restrict__ mask,           // [B][N]
    unsigned short* __restrict__ attnT) {   // [B][N(m)][N(n)]
  const int b = blockIdx.y;
  const int tid = threadIdx.x, lane = tid & 31, w = tid >> 5;
  const int half = lane >> 4, lj = lane & 15;
  const int n0 = blockIdx.x * 128 + w * 16;  // wave's 16 rows
  const unsigned short* q = qT + (size_t)b * Nn * Hh;
  const int* mb = mask + (size_t)b * Nn;

  Frag fa[2];
  {
    const unsigned short* arow = q + (size_t)(n0 + lj) * Hh;
    load_fragA(fa[0], arow + 0, half);
    load_fragA(fa[1], arow + 32, half);
  }
  int   rm[8];
  float Mr[8], Lr[8];
  for (int r = 0; r < 8; ++r) {
    rm[r] = mb[n0 + 8 * half + r];
    Mr[r] = -3.0e38f; Lr[r] = 0.0f;
  }
  // ---- sweep 1: per-lane online row max / row sum (no shuffles in loop) ----
#pragma unroll 2
  for (int mt = 0; mt < Nn / 16; ++mt) {
    const int m = mt * 16 + lj;
    const int cm = mb[m];
    v8f acc; for (int i = 0; i < 8; ++i) acc[i] = 0.0f;
    const unsigned short* brow = q + (size_t)m * Hh;
    Frag fb;
    load_fragB(fb, brow + 0, half);  acc = wmma_bf16(fa[0], fb, acc);
    load_fragB(fb, brow + 32, half); acc = wmma_bf16(fa[1], fb, acc);
    for (int r = 0; r < 8; ++r) {
      float s = (rm[r] && cm) ? acc[r] : NEGV;
      float nM = fmaxf(Mr[r], s);
      Lr[r] = Lr[r] * __expf(Mr[r] - nM) + __expf(s - nM);
      Mr[r] = nM;
    }
  }
  // ---- one-time 16-lane (max,sum) merge per row ----
  for (int r = 0; r < 8; ++r) {
    for (int off = 1; off < 16; off <<= 1) {
      float oM = __shfl_xor(Mr[r], off, 32);
      float oL = __shfl_xor(Lr[r], off, 32);
      float nM = fmaxf(Mr[r], oM);
      Lr[r] = Lr[r] * __expf(Mr[r] - nM) + oL * __expf(oM - nM);
      Mr[r] = nM;
    }
  }
  float iL[8];
  for (int r = 0; r < 8; ++r) iL[r] = 1.0f / (Lr[r] + 1e-30f);
  // ---- sweep 2: recompute, normalize, emit transposed bf16 ----
#pragma unroll 2
  for (int mt = 0; mt < Nn / 16; ++mt) {
    const int m = mt * 16 + lj;
    const int cm = mb[m];
    v8f acc; for (int i = 0; i < 8; ++i) acc[i] = 0.0f;
    const unsigned short* brow = q + (size_t)m * Hh;
    Frag fb;
    load_fragB(fb, brow + 0, half);  acc = wmma_bf16(fa[0], fb, acc);
    load_fragB(fb, brow + 32, half); acc = wmma_bf16(fa[1], fb, acc);
    uint4 pk; unsigned short* tp = (unsigned short*)&pk;
    for (int r = 0; r < 8; ++r) {
      float s = (rm[r] && cm) ? acc[r] : NEGV;
      tp[r] = f2bf(__expf(s - Mr[r]) * iL[r]);
    }
    *(uint4*)(attnT + ((size_t)b * Nn + m) * Nn + n0 + 8 * half) = pk;
  }
}

// colsum[b][m] = sum_n attnT[b][m][n]  (contiguous rows, one warp per row)
__global__ __launch_bounds__(256) void k_colsum(
    const unsigned short* __restrict__ attnT, float* __restrict__ colsum) {
  const int gw = blockIdx.x * 8 + (threadIdx.x >> 5);
  const int lane = threadIdx.x & 31;
  const unsigned short* row = attnT + (size_t)gw * Nn;
  float s = 0.0f;
  for (int i = lane * 8; i < Nn; i += 32 * 8) {
    uint4 v = *(const uint4*)(row + i);
    const unsigned short* tp = (const unsigned short*)&v;
    for (int j = 0; j < 8; ++j) s += bf2f(tp[j]);
  }
  for (int off = 16; off; off >>= 1) s += __shfl_xor(s, off, 32);
  if (lane == 0) colsum[gw] = s;
}

// ---------------------------------------------------------------------------
// x_r GEMM: acc[c][m] = sum_n xv[b][c][n]*attnT[b][m][n]; epilogue applies the
// L1 column renorm (1/(1e-9+colsum[m])) and emits dB = bf16(x - x_r).
// attnT is already [col][k]: both tiles are straight copies -> fully async,
// double-buffered (GLOBAL_LOAD_ASYNC_TO_LDS_B128 + staged s_wait_asynccnt).
// ---------------------------------------------------------------------------
__global__ __launch_bounds__(256) void k_gemm2(
    const unsigned short* __restrict__ xvp, const unsigned short* __restrict__ attnT,
    const float* __restrict__ colsum, const float* __restrict__ xf,
    unsigned short* __restrict__ dB) {
  __shared__ __align__(16) unsigned short ldsA[2][BM_T * BK_T];
  __shared__ __align__(16) unsigned short ldsBT[2][BN_T * BK_T];
  const int b = blockIdx.z, n0 = blockIdx.x * BN_T, m0 = blockIdx.y * BM_T;
  const int tid = threadIdx.x, lane = tid & 31, w = tid >> 5;
  const int wm = w >> 1, wn = w & 1, half = lane >> 4, lj = lane & 15;
  const unsigned short* Ab = xvp + (size_t)b * Cc * Nn;
  const unsigned short* Tb = attnT + (size_t)b * Nn * Nn;

  const int am = tid >> 2, akk = (tid & 3) * 8;   // A: 16B per thread
  const int bm = tid >> 1, bcc = (tid & 1) * 16;  // B: 32B per thread
  const unsigned short* Arow = Ab + (size_t)(m0 + am) * Nn + akk;
  const unsigned short* Brow = Tb + (size_t)(n0 + bm) * Nn + bcc;

  v8f acc[4];
  for (int t = 0; t < 4; ++t)
    for (int i = 0; i < 8; ++i) acc[t][i] = 0.0f;

  // 3 async copies per thread per tile
  auto stage = [&](int k0, int buf) {
    async_lds_b128(ldsA[buf] + am * BK_T + akk, Arow + k0);
    async_lds_b128(ldsBT[buf] + bm * BK_T + bcc, Brow + k0);
    async_lds_b128(ldsBT[buf] + bm * BK_T + bcc + 8, Brow + k0 + 8);
  };
  stage(0, 0);

  for (int k0 = 0; k0 < Nn; k0 += BK_T) {
    const int buf = (k0 / BK_T) & 1;
    const bool more = (k0 + BK_T) < Nn;
    if (more) stage(k0 + BK_T, buf ^ 1);
    if (more) async_wait3(); else async_wait0();
    __syncthreads();
    Frag fa; load_fragA(fa, ldsA[buf] + (wm * 16 + lj) * BK_T, half);
    for (int t = 0; t < 4; ++t) {
      Frag fb; load_fragB(fb, ldsBT[buf] + (wn * 64 + t * 16 + lj) * BK_T, half);
      acc[t] = wmma_bf16(fa, fb, acc[t]);
    }
    __syncthreads();
  }
  for (int t = 0; t < 4; ++t) {
    int col = n0 + wn * 64 + t * 16 + lj;
    float invd = 1.0f / (1e-9f + colsum[b * Nn + col]);
    for (int r = 0; r < 8; ++r) {
      int row = m0 + wm * 16 + 8 * half + r;
      size_t idx = ((size_t)b * Cc + row) * Nn + col;
      dB[idx] = f2bf(xf[idx] - acc[t][r] * invd);
    }
  }
}

// Per-channel sum / sumsq over [B,*,N] (one block per channel, no atomics)
__global__ __launch_bounds__(256) void k_bn_stats(
    const float* __restrict__ t, float* __restrict__ stats) {
  const int c = blockIdx.x, tid = threadIdx.x;
  float s = 0.0f, q = 0.0f;
  for (int b = 0; b < Bn; ++b) {
    const float* p = t + ((size_t)b * Cc + c) * Nn;
    for (int i = tid; i < Nn; i += 256) { float v = p[i]; s += v; q += v * v; }
  }
  __shared__ float ss[256], sq[256];
  ss[tid] = s; sq[tid] = q; __syncthreads();
  for (int off = 128; off; off >>= 1) {
    if (tid < off) { ss[tid] += ss[tid + off]; sq[tid] += sq[tid + off]; }
    __syncthreads();
  }
  if (tid == 0) { stats[c] = ss[0]; stats[Cc + c] = sq[0]; }
}

__device__ __forceinline__ float bn_val(const float* stats, const float* g,
                                        const float* bt, int c, float y) {
  const float inv = 1.0f / (float)(Bn * Nn);
  float m = stats[c] * inv;
  float v = stats[Cc + c] * inv - m * m;
  return (y - m) * rsqrtf(v + 1e-5f) * g[c] + bt[c];
}

// x = relu(bn(y)) ; also emit bf16 copy
__global__ __launch_bounds__(256) void k_bn_relu(
    const float* __restrict__ y, const float* __restrict__ stats,
    const float* __restrict__ g, const float* __restrict__ bt,
    float* __restrict__ xo, unsigned short* __restrict__ xb) {
  const size_t tot = (size_t)Bn * Cc * Nn;
  for (size_t i = (size_t)blockIdx.x * blockDim.x + threadIdx.x; i < tot;
       i += (size_t)gridDim.x * blockDim.x) {
    int c = (int)((i / Nn) % Cc);
    float v = fmaxf(bn_val(stats, g, bt, c, y[i]), 0.0f);
    xo[i] = v; xb[i] = f2bf(v);
  }
}

// x += relu(bn(t)) ; emit bf16 copy + write the concat slice of d_out
__global__ __launch_bounds__(256) void k_bn_res(
    const float* __restrict__ t, const float* __restrict__ stats,
    const float* __restrict__ g, const float* __restrict__ bt,
    float* __restrict__ x, unsigned short* __restrict__ xb,
    float* __restrict__ out, int layer) {
  const size_t tot = (size_t)Bn * Cc * Nn;
  for (size_t i = (size_t)blockIdx.x * blockDim.x + threadIdx.x; i < tot;
       i += (size_t)gridDim.x * blockDim.x) {
    int c = (int)((i / Nn) % Cc);
    float v = x[i] + fmaxf(bn_val(stats, g, bt, c, t[i]), 0.0f);
    x[i] = v; xb[i] = f2bf(v);
    size_t b = i / ((size_t)Cc * Nn);
    size_t rem = i - b * (size_t)Cc * Nn;
    out[b * (size_t)4 * Cc * Nn + (size_t)layer * Cc * Nn + rem] = v;
  }
}

__global__ __launch_bounds__(256) void k_cvt(
    const float* __restrict__ a, unsigned short* __restrict__ o, size_t n) {
  for (size_t i = (size_t)blockIdx.x * blockDim.x + threadIdx.x; i < n;
       i += (size_t)gridDim.x * blockDim.x)
    o[i] = f2bf(a[i]);
}

// ---------------------------------------------------------------------------
extern "C" void kernel_launch(void* const* d_in, const int* in_sizes, int n_in,
                              void* d_out, int out_size, void* d_ws, size_t ws_size,
                              hipStream_t stream) {
  (void)in_sizes; (void)n_in; (void)out_size; (void)ws_size;
  const float* x_in   = (const float*)d_in[0];
  const int*   mask   = (const int*)d_in[1];
  const float* conv1w = (const float*)d_in[2];
  const float* conv2w = (const float*)d_in[3];
  const float* bn1g   = (const float*)d_in[4];
  const float* bn1b   = (const float*)d_in[5];
  const float* bn2g   = (const float*)d_in[6];
  const float* bn2b   = (const float*)d_in[7];
  const float* qkw    = (const float*)d_in[8];
  const float* vw     = (const float*)d_in[9];
  const float* vb     = (const float*)d_in[10];
  const float* tw     = (const float*)d_in[11];
  // d_in[12] = sa_t_b: a per-channel bias is exactly cancelled by train-mode BN.
  const float* sabng  = (const float*)d_in[13];
  const float* sabnb  = (const float*)d_in[14];
  float* out = (float*)d_out;

  // workspace carve-out (~260 MB, attnT dominates and fits the 192MB L2 hot set)
  char* ws = (char*)d_ws; size_t off = 0;
  auto carve = [&](size_t bytes) -> char* {
    char* p = ws + off; off = (off + bytes + 255) & ~(size_t)255; return p;
  };
  const size_t E = (size_t)Bn * Cc * Nn;
  float*          xf     = (float*)carve(E * 4);
  float*          yf     = (float*)carve(E * 4);
  unsigned short* xb     = (unsigned short*)carve(E * 2);
  unsigned short* xv     = (unsigned short*)carve(E * 2);
  unsigned short* dB     = (unsigned short*)carve(E * 2);
  unsigned short* qT     = (unsigned short*)carve((size_t)Bn * Nn * Hh * 2);
  unsigned short* attnT  = (unsigned short*)carve((size_t)Bn * Nn * Nn * 2);
  float*          colsum = (float*)carve((size_t)Bn * Nn * 4);
  float*          stats  = (float*)carve((size_t)2 * Cc * 4);
  unsigned short* wbA    = (unsigned short*)carve((size_t)Cc * Cc * 2);
  unsigned short* wbB    = (unsigned short*)carve((size_t)Cc * Cc * 2);
  unsigned short* wbQ    = (unsigned short*)carve((size_t)Hh * Cc * 2);

  const dim3 blk(256);
  const dim3 gg(Nn / BN_T, Cc / BM_T, Bn);   // 256-wide GEMMs
  const dim3 gq(Nn / BN_T, 1, Bn);           // M=64 (q) GEMM

  // ---- stem: conv1 -> bn -> relu -> conv2 -> bn -> relu ----
  k_cvt<<<dim3(512), blk, 0, stream>>>(x_in, xb, E);
  k_cvt<<<dim3(64), blk, 0, stream>>>(conv1w, wbA, (size_t)Cc * Cc);
  k_gemm<0><<<gg, blk, 0, stream>>>(wbA, xb, Cc, Cc, Nn, nullptr, yf, nullptr);
  k_bn_stats<<<dim3(Cc), blk, 0, stream>>>(yf, stats);
  k_bn_relu<<<dim3(1024), blk, 0, stream>>>(yf, stats, bn1g, bn1b, xf, xb);
  k_cvt<<<dim3(64), blk, 0, stream>>>(conv2w, wbA, (size_t)Cc * Cc);
  k_gemm<0><<<gg, blk, 0, stream>>>(wbA, xb, Cc, Cc, Nn, nullptr, yf, nullptr);
  k_bn_stats<<<dim3(Cc), blk, 0, stream>>>(yf, stats);
  k_bn_relu<<<dim3(1024), blk, 0, stream>>>(yf, stats, bn2g, bn2b, xf, xb);

  // ---- 4 self-attention layers ----
  for (int L = 0; L < 4; ++L) {
    k_cvt<<<dim3(16), blk, 0, stream>>>(qkw + (size_t)L * Hh * Cc, wbQ, (size_t)Hh * Cc);
    k_cvt<<<dim3(64), blk, 0, stream>>>(vw + (size_t)L * Cc * Cc, wbA, (size_t)Cc * Cc);
    k_cvt<<<dim3(64), blk, 0, stream>>>(tw + (size_t)L * Cc * Cc, wbB, (size_t)Cc * Cc);

    k_gemm<2><<<gq, blk, 0, stream>>>(wbQ, xb, Hh, Cc, Nn, nullptr, nullptr, qT);
    k_gemm<1><<<gg, blk, 0, stream>>>(wbA, xb, Cc, Cc, Nn, vb + (size_t)L * Cc, nullptr, xv);
    k_energy<<<dim3(Nn / 128, Bn), blk, 0, stream>>>(qT, mask, attnT);
    k_colsum<<<dim3(Bn * Nn / 8), blk, 0, stream>>>(attnT, colsum);
    k_gemm2<<<gg, blk, 0, stream>>>(xv, attnT, colsum, xf, dB);
    k_gemm<0><<<gg, blk, 0, stream>>>(wbB, dB, Cc, Cc, Nn, nullptr, yf, nullptr);
    k_bn_stats<<<dim3(Cc), blk, 0, stream>>>(yf, stats);
    k_bn_res<<<dim3(1024), blk, 0, stream>>>(yf, stats, sabng + (size_t)L * Cc,
                                             sabnb + (size_t)L * Cc, xf, xb, out, L);
  }
}
